// GINModel_12137577578701
// MI455X (gfx1250) — compile-verified, matrix operand order checked by reading, not run on hardware
//
#include <hip/hip_runtime.h>

typedef __bf16 bf16_t;
typedef __attribute__((ext_vector_type(8)))  __bf16 v8bf;
typedef __attribute__((ext_vector_type(16))) __bf16 v16bf;
typedef __attribute__((ext_vector_type(8)))  float  v8f;

namespace {
constexpr int kN  = 100000;   // nodes (6250 * 16 -> 3125 M-tile pairs)
constexpr int kE  = 400000;   // edges
constexpr int kG  = 4096;     // graphs (256 * 16)
constexpr int kD  = 300;      // hidden dim
constexpr int kL  = 5;        // layers
constexpr int kTasks = 12;
constexpr int kD2 = 600;      // 2*D
constexpr int kKP  = 320;     // K-pad for 300 (multiple of 32)
constexpr int kNP1 = 608;     // N-pad for 600 (38 tiles, even; also K for gemm2)
constexpr int kNP2 = 320;     // N-pad for 300 (20 tiles, even -> no tile guard)
constexpr int kF = 512;
constexpr int kP = 256;
constexpr float kEps = 1e-5f;
}

// 16-bit A-fragment (16x32): both per-lane halves are contiguous 8-element
// (16B) runs -> two b128 loads per fragment.
__device__ __forceinline__ v16bf load_frag(const bf16_t* __restrict__ p)
{
    v8bf lo = *reinterpret_cast<const v8bf*>(p);
    v8bf hi = *reinterpret_cast<const v8bf*>(p + 16);
    return __builtin_shufflevector(lo, hi,
             0,1,2,3,4,5,6,7,8,9,10,11,12,13,14,15);
}

__device__ __forceinline__ v8f wmma_bf16(v16bf a, v16bf b, v8f c)
{
    return __builtin_amdgcn_wmma_f32_16x16x32_bf16(
               false, a, false, b, (short)0, c, false, false);
}

// ---------------------------------------------------------------------------
// bf16 WMMA GEMM: C[M,N] = A[M,K] * Bt[N,K]^T (+bias, optional relu)
// A: bf16 row-major, lda = padded K (multiple of 32, rows 16B aligned)
// Bt: weight stored transposed [Npad, Kpad]
// Each wave computes a 32x32 output block (2x2 16x16 tiles) with fragment
// reuse; K loop is software-pipelined (double-buffered fragments).
// Requires Mtiles and Ntiles even (caller pads), so no EXEC divergence in
// the main loop and WMMA always runs with EXEC all-1s.
// ---------------------------------------------------------------------------
__global__ void gemm_bf16_wmma(const bf16_t* __restrict__ A, int lda,
                               const bf16_t* __restrict__ Bt, int ldb,
                               const float* __restrict__ bias,
                               float* __restrict__ Cf, int ldcf,
                               bf16_t* __restrict__ Cbf, int ldcbf,
                               int Mtiles, int Ntiles, int K, int Nout, int doRelu)
{
    const int wave = blockIdx.x * ((int)blockDim.x >> 5) + ((int)threadIdx.x >> 5);
    const int npairs = Ntiles >> 1;
    const int mpairs = Mtiles >> 1;
    if (wave >= mpairs * npairs) return;        // wave-uniform exit
    const int mp = wave / npairs;
    const int np = wave - mp * npairs;
    const int lane = (int)threadIdx.x & 31;
    const int hs = (lane >> 4) & 1;
    const int r  = lane & 15;
    const int m0 = mp << 5;
    const int n0 = np << 5;

    const bf16_t* Ap0 = A  + (size_t)(m0 + r)      * lda + hs * 8;
    const bf16_t* Ap1 = A  + (size_t)(m0 + 16 + r) * lda + hs * 8;
    const bf16_t* Bp0 = Bt + (size_t)(n0 + r)      * ldb + hs * 8;
    const bf16_t* Bp1 = Bt + (size_t)(n0 + 16 + r) * ldb + hs * 8;

    v8f c00 = {}, c01 = {}, c10 = {}, c11 = {};

    v16bf a0 = load_frag(Ap0);
    v16bf a1 = load_frag(Ap1);
    v16bf b0 = load_frag(Bp0);
    v16bf b1 = load_frag(Bp1);

    for (int k = 32; k < K; k += 32) {
        // issue next-step loads before consuming current fragments
        v16bf na0 = load_frag(Ap0 + k);
        v16bf na1 = load_frag(Ap1 + k);
        v16bf nb0 = load_frag(Bp0 + k);
        v16bf nb1 = load_frag(Bp1 + k);
        __builtin_prefetch(Ap0 + k + 64, 0, 1);   // global_prefetch_b8
        __builtin_prefetch(Ap1 + k + 64, 0, 1);
        c00 = wmma_bf16(a0, b0, c00);
        c01 = wmma_bf16(a0, b1, c01);
        c10 = wmma_bf16(a1, b0, c10);
        c11 = wmma_bf16(a1, b1, c11);
        a0 = na0; a1 = na1; b0 = nb0; b1 = nb1;
    }
    c00 = wmma_bf16(a0, b0, c00);
    c01 = wmma_bf16(a0, b1, c01);
    c10 = wmma_bf16(a1, b0, c10);
    c11 = wmma_bf16(a1, b1, c11);

    // C/D layout: VGPR e -> row tileRow0 + e + hs*8 ; col tileCol0 + (lane&15)
    const int cn0 = n0 + r;
    const int cn1 = n0 + 16 + r;
    const float bias0 = (cn0 < Nout) ? bias[cn0] : 0.0f;
    const float bias1 = (cn1 < Nout) ? bias[cn1] : 0.0f;

    auto emit = [&](int m, int n, float bn, float acc) {
        float v = acc + bn;
        if (doRelu && v < 0.0f) v = 0.0f;
        if (Cbf) Cbf[(size_t)m * ldcbf + n] = (n < Nout) ? (bf16_t)v : (bf16_t)0.0f;
        if (Cf && n < Nout) Cf[(size_t)m * ldcf + n] = v;
    };

#pragma unroll
    for (int e = 0; e < 8; ++e) {
        const int mA = m0 + e + hs * 8;
        const int mB = mA + 16;
        emit(mA, cn0, bias0, c00[e]);
        emit(mA, cn1, bias1, c01[e]);
        emit(mB, cn0, bias0, c10[e]);
        emit(mB, cn1, bias1, c11[e]);
    }
}

// --------------------------- support kernels -------------------------------

__global__ void node_embed_kernel(const int* __restrict__ atom,
                                  const int* __restrict__ chir,
                                  const float* __restrict__ e1,
                                  const float* __restrict__ e2,
                                  float* __restrict__ h)
{
    const int n = blockIdx.x;
    const int a = atom[n], c = chir[n];
    for (int d = threadIdx.x; d < kD; d += blockDim.x)
        h[(size_t)n * kD + d] = e1[(size_t)a * kD + d] + e2[(size_t)c * kD + d];
}

__global__ void copy_f32(const float* __restrict__ src, float* __restrict__ dst, size_t n)
{
    const size_t stride = (size_t)gridDim.x * blockDim.x;
    for (size_t i = (size_t)blockIdx.x * blockDim.x + threadIdx.x; i < n; i += stride)
        dst[i] = src[i];
}

__global__ void zero_f32(float* __restrict__ dst, size_t n)
{
    const size_t stride = (size_t)gridDim.x * blockDim.x;
    for (size_t i = (size_t)blockIdx.x * blockDim.x + threadIdx.x; i < n; i += stride)
        dst[i] = 0.0f;
}

// one block per edge: agg[row] += h[col] + ee1[type] + ee2[dir]
__global__ void edge_msg_kernel(const int* __restrict__ erow, const int* __restrict__ ecol,
                                const int* __restrict__ etype, const int* __restrict__ edir,
                                const float* __restrict__ ee1, const float* __restrict__ ee2,
                                const float* __restrict__ h, float* __restrict__ agg)
{
    const int e = blockIdx.x;
    const int rr = erow[e], cc = ecol[e], t = etype[e], dd = edir[e];
    const float* hs  = h   + (size_t)cc * kD;
    const float* t1  = ee1 + (size_t)t  * kD;
    const float* t2  = ee2 + (size_t)dd * kD;
    float* dst = agg + (size_t)rr * kD;
    for (int d = threadIdx.x; d < kD; d += blockDim.x)
        atomicAdd(&dst[d], hs[d] + t1[d] + t2[d]);
}

// f32 [rows, D0] -> bf16 [rows, Kpad], zero-filled pad columns
__global__ void convert_a_pad(const float* __restrict__ X, bf16_t* __restrict__ Xb,
                              int rows, int D0, int Kpad)
{
    const size_t total = (size_t)rows * Kpad;
    const size_t stride = (size_t)gridDim.x * blockDim.x;
    for (size_t i = (size_t)blockIdx.x * blockDim.x + threadIdx.x; i < total; i += stride) {
        const int k = (int)(i % Kpad);
        const size_t rrow = i / Kpad;
        Xb[i] = (k < D0) ? (bf16_t)X[rrow * D0 + k] : (bf16_t)0.0f;
    }
}

// f32 W[K0, N0] row-major -> bf16 Wt[Npad, Kpad] (transposed, zero-padded)
__global__ void convert_w_transpose(const float* __restrict__ W, bf16_t* __restrict__ Wt,
                                    int K0, int N0, int Kpad, int Npad)
{
    const size_t total = (size_t)Npad * Kpad;
    const size_t stride = (size_t)gridDim.x * blockDim.x;
    for (size_t i = (size_t)blockIdx.x * blockDim.x + threadIdx.x; i < total; i += stride) {
        const int k = (int)(i % Kpad);
        const int n = (int)(i / Kpad);
        float v = (n < N0 && k < K0) ? W[(size_t)k * N0 + n] : 0.0f;
        Wt[i] = (bf16_t)v;
    }
}

// per-column sum / sum-of-squares partials -> atomics into stats[2*D]
__global__ void bn_stats_kernel(const float* __restrict__ z, float* __restrict__ stats, int rows)
{
    const int d = threadIdx.x;
    if (d >= kD) return;
    float s = 0.0f, sq = 0.0f;
    for (int r = blockIdx.x; r < rows; r += gridDim.x) {
        const float v = z[(size_t)r * kD + d];
        s += v; sq += v * v;
    }
    atomicAdd(&stats[d], s);
    atomicAdd(&stats[kD + d], sq);
}

__global__ void bn_apply_kernel(const float* __restrict__ z, const float* __restrict__ stats,
                                const float* __restrict__ gamma, const float* __restrict__ beta,
                                float* __restrict__ h, size_t total)
{
    const float invN = 1.0f / (float)kN;
    const size_t stride = (size_t)gridDim.x * blockDim.x;
    for (size_t i = (size_t)blockIdx.x * blockDim.x + threadIdx.x; i < total; i += stride) {
        const int d = (int)(i % kD);
        const float mean = stats[d] * invN;
        const float var  = stats[kD + d] * invN - mean * mean;
        float v = gamma[d] * (z[i] - mean) * rsqrtf(var + kEps) + beta[d];
        h[i] = (v > 0.0f) ? v : 0.0f;
    }
}

// one block per node: pooled[batch[n]] += h[n]
__global__ void pool_kernel(const int* __restrict__ batch, const float* __restrict__ h,
                            float* __restrict__ pooled)
{
    const int n = blockIdx.x;
    const int g = batch[n];
    const float* src = h + (size_t)n * kD;
    float* dst = pooled + (size_t)g * kD;
    for (int d = threadIdx.x; d < kD; d += blockDim.x)
        atomicAdd(&dst[d], src[d]);
}

// out[g,t] = sum_k f3[g,k] * w3[k,t] + b3[t]
__global__ void head_final_kernel(const bf16_t* __restrict__ f3,
                                  const float* __restrict__ w3,
                                  const float* __restrict__ b3,
                                  float* __restrict__ out)
{
    const int g = blockIdx.x;
    const int t = threadIdx.x;
    if (t >= kTasks) return;
    float acc = b3[t];
    const bf16_t* fr = f3 + (size_t)g * kP;
    for (int k = 0; k < kP; ++k)
        acc += (float)fr[k] * w3[(size_t)k * kTasks + t];
    out[(size_t)g * kTasks + t] = acc;
}

// ------------------------------- launch ------------------------------------

static void launch_gemm(const bf16_t* A, int lda, const bf16_t* Bt, int ldb,
                        const float* bias, float* Cf, int ldcf, bf16_t* Cbf, int ldcbf,
                        int Mtiles, int Ntiles, int K, int Nout, int relu, hipStream_t s)
{
    const long long waves = (long long)(Mtiles / 2) * (Ntiles / 2);
    const int blocks = (int)((waves + 7) / 8);   // 256 threads = 8 waves/block
    gemm_bf16_wmma<<<blocks, 256, 0, s>>>(A, lda, Bt, ldb, bias, Cf, ldcf, Cbf, ldcbf,
                                          Mtiles, Ntiles, K, Nout, relu);
}

extern "C" void kernel_launch(void* const* d_in, const int* in_sizes, int n_in,
                              void* d_out, int out_size, void* d_ws, size_t ws_size,
                              hipStream_t stream)
{
    (void)in_sizes; (void)n_in; (void)out_size; (void)ws_size;

    const int*   node_atom = (const int*)d_in[0];
    const int*   node_chir = (const int*)d_in[1];
    const int*   e_row     = (const int*)d_in[2];
    const int*   e_col     = e_row + kE;
    const int*   e_type    = (const int*)d_in[3];
    const int*   e_dir     = (const int*)d_in[4];
    const int*   batch     = (const int*)d_in[5];
    const float* x_emb1    = (const float*)d_in[6];
    const float* x_emb2    = (const float*)d_in[7];
    const float* edge_emb1 = (const float*)d_in[8];
    const float* edge_emb2 = (const float*)d_in[9];
    const float* mlp_w1    = (const float*)d_in[10];
    const float* mlp_b1    = (const float*)d_in[11];
    const float* mlp_w2    = (const float*)d_in[12];
    const float* mlp_b2    = (const float*)d_in[13];
    const float* bn_g      = (const float*)d_in[14];
    const float* bn_b      = (const float*)d_in[15];
    const float* feat_w    = (const float*)d_in[16];
    const float* feat_b    = (const float*)d_in[17];
    const float* ph_w1     = (const float*)d_in[18];
    const float* ph_b1     = (const float*)d_in[19];
    const float* ph_w2     = (const float*)d_in[20];
    const float* ph_b2     = (const float*)d_in[21];
    const float* ph_w3     = (const float*)d_in[22];
    const float* ph_b3     = (const float*)d_in[23];
    float* out = (float*)d_out;

    // workspace carve (256B aligned)
    char* ws = (char*)d_ws;
    size_t off = 0;
    auto carve = [&](size_t bytes) -> void* {
        void* p = ws + off;
        off = (off + bytes + 255) & ~(size_t)255;
        return p;
    };
    float*  h      = (float*) carve((size_t)kN * kD * 4);
    float*  agg    = (float*) carve((size_t)kN * kD * 4);   // also reused as z
    bf16_t* abf    = (bf16_t*)carve((size_t)kN * kKP * 2);
    bf16_t* a2bf   = (bf16_t*)carve((size_t)kN * kNP1 * 2);
    bf16_t* w1t    = (bf16_t*)carve((size_t)kNP1 * kKP * 2);
    bf16_t* w2t    = (bf16_t*)carve((size_t)kNP2 * kNP1 * 2);
    bf16_t* wft    = (bf16_t*)carve((size_t)kF * kKP * 2);
    bf16_t* wp1t   = (bf16_t*)carve((size_t)kP * kF * 2);
    bf16_t* wp2t   = (bf16_t*)carve((size_t)kP * kP * 2);
    float*  stats  = (float*) carve((size_t)2 * kD * 4);
    float*  pooled = (float*) carve((size_t)kG * kD * 4);
    bf16_t* pbf    = (bf16_t*)carve((size_t)kG * kKP * 2);
    bf16_t* f1     = (bf16_t*)carve((size_t)kG * kF * 2);
    bf16_t* f2     = (bf16_t*)carve((size_t)kG * kP * 2);
    bf16_t* f3     = (bf16_t*)carve((size_t)kG * kP * 2);

    // h = x_emb1[atom] + x_emb2[chir]
    node_embed_kernel<<<kN, 256, 0, stream>>>(node_atom, node_chir, x_emb1, x_emb2, h);

    for (int l = 0; l < kL; ++l) {
        convert_w_transpose<<<1024, 256, 0, stream>>>(mlp_w1 + (size_t)l * kD * kD2,
                                                      w1t, kD, kD2, kKP, kNP1);
        convert_w_transpose<<<1024, 256, 0, stream>>>(mlp_w2 + (size_t)l * kD2 * kD,
                                                      w2t, kD2, kD, kNP1, kNP2);
        // agg = h (self connection), then scatter-add messages
        copy_f32<<<2048, 256, 0, stream>>>(h, agg, (size_t)kN * kD);
        edge_msg_kernel<<<kE, 256, 0, stream>>>(e_row, e_col, e_type, e_dir,
                                                edge_emb1 + (size_t)l * 5 * kD,
                                                edge_emb2 + (size_t)l * 3 * kD, h, agg);
        // z1 = relu(agg @ W1 + b1)   [N, 600] -> bf16 padded [N, 608]
        convert_a_pad<<<2048, 256, 0, stream>>>(agg, abf, kN, kD, kKP);
        launch_gemm(abf, kKP, w1t, kKP, mlp_b1 + (size_t)l * kD2,
                    nullptr, 0, a2bf, kNP1,
                    kN / 16, kNP1 / 16, kKP, kD2, 1, stream);
        // z = z1 @ W2 + b2   [N, 300] fp32 (reuse agg buffer)
        launch_gemm(a2bf, kNP1, w2t, kNP1, mlp_b2 + (size_t)l * kD,
                    agg, kD, nullptr, 0,
                    kN / 16, kNP2 / 16, kNP1, kD, 0, stream);
        // BatchNorm (batch stats) + relu -> h
        zero_f32<<<8, 256, 0, stream>>>(stats, (size_t)2 * kD);
        bn_stats_kernel<<<512, 320, 0, stream>>>(agg, stats, kN);
        bn_apply_kernel<<<2048, 256, 0, stream>>>(agg, stats, bn_g + (size_t)l * kD,
                                                  bn_b + (size_t)l * kD, h, (size_t)kN * kD);
    }

    // global_add_pool
    zero_f32<<<2048, 256, 0, stream>>>(pooled, (size_t)kG * kD);
    pool_kernel<<<kN, 256, 0, stream>>>(batch, h, pooled);

    // head: 300 -> 512 -> 256 -> 256 -> 12
    convert_a_pad<<<512, 256, 0, stream>>>(pooled, pbf, kG, kD, kKP);
    convert_w_transpose<<<512, 256, 0, stream>>>(feat_w, wft, kD, kF, kKP, kF);
    convert_w_transpose<<<512, 256, 0, stream>>>(ph_w1, wp1t, kF, kP, kF, kP);
    convert_w_transpose<<<256, 256, 0, stream>>>(ph_w2, wp2t, kP, kP, kP, kP);
    launch_gemm(pbf, kKP, wft, kKP, feat_b, nullptr, 0, f1, kF,
                kG / 16, kF / 16, kKP, kF, 1, stream);
    launch_gemm(f1, kF, wp1t, kF, ph_b1, nullptr, 0, f2, kP,
                kG / 16, kP / 16, kF, kP, 1, stream);
    launch_gemm(f2, kP, wp2t, kP, ph_b2, nullptr, 0, f3, kP,
                kG / 16, kP / 16, kP, kP, 1, stream);
    head_final_kernel<<<kG, 16, 0, stream>>>(f3, ph_w3, ph_b3, out);
}
